// MHA_1906965480193
// MI455X (gfx1250) — matrix operand order repaired and checked by reading, not compiled, not run on hardware
//
#include <hip/hip_runtime.h>

typedef __bf16 bf16_t;
typedef __bf16 v16bf __attribute__((ext_vector_type(16)));
typedef __bf16 v8bf  __attribute__((ext_vector_type(8)));
typedef float  v8f   __attribute__((ext_vector_type(8)));

#define BDIM 4
#define TDIM 2048
#define CDIM 1024
#define NHEAD 16
#define HDIM 64

__device__ inline v8f zero8() {
  v8f z = {0.f,0.f,0.f,0.f,0.f,0.f,0.f,0.f};
  return z;
}

__device__ inline v16bf frag16(const bf16_t* p0, const bf16_t* p1) {
  v8bf lo = *(const v8bf*)p0;
  v8bf hi = *(const v8bf*)p1;
  return __builtin_shufflevector(lo, hi, 0,1,2,3,4,5,6,7,8,9,10,11,12,13,14,15);
}

__device__ inline v8f wmma_bf16(v16bf a, v16bf b, v8f c) {
  // D = A(16x32) * B(32x16) + C, f32 accumulate
  return __builtin_amdgcn_wmma_f32_16x16x32_bf16(false, a, false, b, (short)0, c, false, false);
}

// ---------------------------------------------------------------------------
// Kernel 0: f32 -> bf16 conversion (n multiple of 1024)
// ---------------------------------------------------------------------------
__global__ void cvt_kernel(const float* __restrict__ src, bf16_t* __restrict__ dst, int n) {
  int i = (blockIdx.x * blockDim.x + threadIdx.x) * 4;
  if (i < n) {
    float4 v = *(const float4*)(src + i);
    dst[i + 0] = (bf16_t)v.x;
    dst[i + 1] = (bf16_t)v.y;
    dst[i + 2] = (bf16_t)v.z;
    dst[i + 3] = (bf16_t)v.w;
  }
}

// ---------------------------------------------------------------------------
// Kernel 1: RoPE cos/sin tables: ctab/stab[t*32 + j] = cos/sin(t * base^(-2j/64))
// ---------------------------------------------------------------------------
__global__ void rope_tab_kernel(float* __restrict__ ctab, float* __restrict__ stab) {
  int idx = blockIdx.x * blockDim.x + threadIdx.x;   // 0 .. 65535
  int t = idx >> 5;
  int j = idx & 31;
  float invf = expf(-((float)(2 * j) / 64.0f) * 9.210340371976184f); // ln(10000)
  float ang = (float)t * invf;
  ctab[idx] = cosf(ang);
  stab[idx] = sinf(ang);
}

// ---------------------------------------------------------------------------
// Kernel 2: GEMM  out[m][n] = sum_k A[m][k] * W[n][k]   (M=8192,N=1024,K=1024)
// A, W are bf16 in global; LDS double-buffered via async-to-LDS copies.
// mode 0: plain f32 write  |  mode 1: RoPE + head-split bf16  |  mode 2: head-split bf16
// grid (8, 64), block 256 (8 waves), block tile 128x128, wave tile 32x64
// ---------------------------------------------------------------------------
__global__ __launch_bounds__(256)
void gemm_rope_kernel(const bf16_t* __restrict__ A, const bf16_t* __restrict__ W,
                      float* __restrict__ outF, bf16_t* __restrict__ outBF,
                      const float* __restrict__ ctab, const float* __restrict__ stab,
                      int mode)
{
  // 80B row pitch: 16B-aligned fragment chunks, bank-conflict-free; 2 buffers
  __shared__ __align__(16) bf16_t As[2][128][40];
  __shared__ __align__(16) bf16_t Bs[2][128][40];

  const int tid   = threadIdx.x;
  const int lane  = tid & 31;
  const int wid   = tid >> 5;
  const int waveM = wid >> 1;          // 0..3
  const int waveN = wid & 1;           // 0..1
  const int blockM = blockIdx.y * 128;
  const int blockN = blockIdx.x * 128;
  const int l16   = lane & 15;
  const int lhalf = (lane < 16) ? 0 : 1;

  v8f acc[2][4];
  #pragma unroll
  for (int i = 0; i < 2; ++i)
    #pragma unroll
    for (int j = 0; j < 4; ++j) acc[i][j] = zero8();

  // ---- async copy setup: 512 16B-chunks per 128x32 tile, 2 per thread per tile
  const int r0  = tid >> 2;            // 0..63
  const int kk0 = (tid & 3) * 8;       // bf16 element offset within 32-wide k tile
  const unsigned aoff0 = ((unsigned)(blockM + r0) * CDIM + kk0) * 2u;  // bytes
  const unsigned aoff1 = aoff0 + 64u * CDIM * 2u;                      // row + 64
  const unsigned boff0 = ((unsigned)(blockN + r0) * CDIM + kk0) * 2u;
  const unsigned boff1 = boff0 + 64u * CDIM * 2u;
  const unsigned lA0 = (unsigned)(uintptr_t)&As[0][r0][kk0];
  const unsigned lA1 = (unsigned)(uintptr_t)&As[0][r0 + 64][kk0];
  const unsigned lB0 = (unsigned)(uintptr_t)&Bs[0][r0][kk0];
  const unsigned lB1 = (unsigned)(uintptr_t)&Bs[0][r0 + 64][kk0];
  const unsigned bufstride = 128u * 40u * 2u;   // 10240 bytes

  auto issue_tile = [&](int kt, int buf) {
    const unsigned ko = (unsigned)kt * 64u;     // k-step byte advance
    const unsigned bo = (unsigned)buf * bufstride;
    asm volatile(
      "global_load_async_to_lds_b128 %0, %4, %8\n"
      "global_load_async_to_lds_b128 %1, %5, %8\n"
      "global_load_async_to_lds_b128 %2, %6, %9\n"
      "global_load_async_to_lds_b128 %3, %7, %9\n"
      :
      : "v"(lA0 + bo), "v"(lA1 + bo), "v"(lB0 + bo), "v"(lB1 + bo),
        "v"(aoff0 + ko), "v"(aoff1 + ko), "v"(boff0 + ko), "v"(boff1 + ko),
        "s"(A), "s"(W)
      : "memory");
  };

  issue_tile(0, 0);

  const int kb0 = lhalf ? 8 : 0;
  const int kbb = lhalf ? 16 : 0;

  for (int kt = 0; kt < 32; ++kt) {
    const int buf = kt & 1;
    if (kt < 31) {
      issue_tile(kt + 1, buf ^ 1);
      asm volatile("s_wait_asynccnt 0x4" ::: "memory");  // current tile (4 older ops) landed
    } else {
      asm volatile("s_wait_asynccnt 0x0" ::: "memory");
    }
    __syncthreads();

    // A fragment: lane half selects K sub-range {0..7,16..23} vs {8..15,24..31}
    v16bf afrag[2];
    #pragma unroll
    for (int mt = 0; mt < 2; ++mt) {
      int mrow = waveM * 32 + mt * 16 + l16;
      afrag[mt] = frag16(&As[buf][mrow][kb0], &As[buf][mrow][kb0 + 16]);
    }
    // B fragment: lane half selects K sub-range {0..7,16..23} vs {8..15,24..31}, lane%16 = N
    v16bf bfrag[4];
    #pragma unroll
    for (int nt = 0; nt < 4; ++nt) {
      int nrow = waveN * 64 + nt * 16 + l16;
      bfrag[nt] = frag16(&Bs[buf][nrow][kb0], &Bs[buf][nrow][kb0 + 16]);
    }
    #pragma unroll
    for (int mt = 0; mt < 2; ++mt)
      #pragma unroll
      for (int nt = 0; nt < 4; ++nt)
        acc[mt][nt] = wmma_bf16(afrag[mt], bfrag[nt], acc[mt][nt]);
    __syncthreads();
  }

  // ---- epilogue ----
  if (mode == 0) {
    #pragma unroll
    for (int mt = 0; mt < 2; ++mt) {
      int mbase = blockM + waveM * 32 + mt * 16 + lhalf * 8;
      #pragma unroll
      for (int nt = 0; nt < 4; ++nt) {
        int ng = blockN + waveN * 64 + nt * 16 + l16;
        #pragma unroll
        for (int r = 0; r < 8; ++r)
          outF[(size_t)(mbase + r) * CDIM + ng] = acc[mt][nt][r];
      }
    }
  } else {
    const int hh = (blockN >> 6) + waveN;     // head (wave N-tile == one head)
    #pragma unroll
    for (int mt = 0; mt < 2; ++mt) {
      int mbase = blockM + waveM * 32 + mt * 16 + lhalf * 8;
      #pragma unroll
      for (int nt = 0; nt < 4; ++nt) {
        int d = nt * 16 + l16;                // 0..63 within head
        int j = d & 31;
        #pragma unroll
        for (int r = 0; r < 8; ++r) {
          int mg = mbase + r;
          int b = mg >> 11;
          int t = mg & 2047;
          float val = acc[mt][nt][r];
          if (mode == 1) {
            float c  = ctab[t * 32 + j];
            float s  = stab[t * 32 + j];
            float pv = acc[mt][nt ^ 2][r];    // element d +/- 32: same lane, partner subtile
            float rot = (nt < 2) ? -pv : pv;
            val = val * c + rot * s;
          }
          outBF[(((size_t)b * NHEAD + hh) * TDIM + t) * HDIM + d] = (bf16_t)val;
        }
      }
    }
  }
}

// ---------------------------------------------------------------------------
// Kernel 3: causal flash attention.  grid (T/128, B*H), block 256 (8 waves).
// Wave w owns query rows [i0+16w, i0+16w+16). Key blocks of 64. K tile via async DMA.
// ---------------------------------------------------------------------------
__global__ __launch_bounds__(256)
void flash_kernel(const bf16_t* __restrict__ Q, const bf16_t* __restrict__ K,
                  const bf16_t* __restrict__ V, bf16_t* __restrict__ O)
{
  __shared__ __align__(16) bf16_t smem[18432];           // 36 KB
  bf16_t* Kt  = smem;                                    // 64 x 72 (K tile, [key][d])
  bf16_t* VtT = smem + 64 * 72;                          // 64 x 72 (V tile transposed, [d][key])
  bf16_t* Ps  = smem + 2 * 64 * 72;                      // 8 waves x 16 x 72 (P scratch)
  bf16_t* Qs  = smem;                                    // 128 x 72 (overlaps Kt+VtT, pre-loop only)

  const int tid   = threadIdx.x;
  const int lane  = tid & 31;
  const int wid   = tid >> 5;
  const int l16   = lane & 15;
  const int lhalf = (lane < 16) ? 0 : 1;

  const int bh = blockIdx.y;
  const int b  = bh >> 4;
  const int h  = bh & 15;
  const int i0 = blockIdx.x * 128;

  const bf16_t* qp = Q + (size_t)bh * TDIM * HDIM;
  const bf16_t* kp = K + (size_t)bh * TDIM * HDIM;
  const bf16_t* vp = V + (size_t)bh * TDIM * HDIM;

  // ---- stage Q tile (128 x 64) and pull per-wave A fragments ----
  {
    int rl = tid >> 2;
    int ds = (tid & 3) * 16;
    #pragma unroll
    for (int p = 0; p < 2; ++p) {
      int row = rl + p * 64;
      const v8bf* src = (const v8bf*)(qp + (size_t)(i0 + row) * HDIM + ds);
      *(v8bf*)&Qs[row * 72 + ds]     = src[0];
      *(v8bf*)&Qs[row * 72 + ds + 8] = src[1];
    }
  }
  __syncthreads();
  const int kb0  = lhalf ? 8 : 0;
  const int qrow = wid * 16 + l16;
  v16bf qa0 = frag16(&Qs[qrow * 72 + kb0],      &Qs[qrow * 72 + kb0 + 16]);
  v16bf qa1 = frag16(&Qs[qrow * 72 + 32 + kb0], &Qs[qrow * 72 + 32 + kb0 + 16]);
  __syncthreads();

  v8f o[4];
  #pragma unroll
  for (int i = 0; i < 4; ++i) o[i] = zero8();
  float mrow[8], lrow[8];
  #pragma unroll
  for (int r = 0; r < 8; ++r) { mrow[r] = -3.0e38f; lrow[r] = 0.0f; }

  const int jmaxb = (i0 >> 6) + 1;
  const int kbb   = lhalf ? 16 : 0;

  // per-thread 16B chunk for async K copy
  const int krl = tid >> 2;                   // key row 0..63
  const int kds = (tid & 3) * 16;             // d start
  const unsigned kvoff = ((unsigned)krl * HDIM + kds) * 2u;            // bytes
  const unsigned kldsa = (unsigned)(uintptr_t)&Kt[krl * 72 + kds];

  for (int jb = 0; jb <= jmaxb; ++jb) {
    const int jbase = jb << 6;
    // async DMA: K tile global -> LDS (pure bf16 copy)
    asm volatile("global_load_async_to_lds_b128 %0, %1, %2"
                 :
                 : "v"(kldsa), "v"(kvoff + (unsigned)jbase * HDIM * 2u), "s"(kp)
                 : "memory");
    // manual transpose for V tile
    {
      const v8bf* vsrc = (const v8bf*)(vp + (size_t)(jbase + krl) * HDIM + kds);
      v8bf v0 = vsrc[0], v1 = vsrc[1];
      #pragma unroll
      for (int e = 0; e < 8; ++e) VtT[(kds + e) * 72 + krl]     = v0[e];
      #pragma unroll
      for (int e = 0; e < 8; ++e) VtT[(kds + 8 + e) * 72 + krl] = v1[e];
    }
    asm volatile("s_wait_asynccnt 0x0" ::: "memory");
    __syncthreads();

    if (jbase <= i0 + wid * 16 + 15) {   // wave-uniform: EXEC stays all-ones
      // S = Q * K^T (scaled later)
      v8f s[4];
      #pragma unroll
      for (int nt = 0; nt < 4; ++nt) {
        const bf16_t* kr = &Kt[(nt * 16 + l16) * 72];
        v16bf kf0 = frag16(kr + kb0,      kr + kb0 + 16);
        v16bf kf1 = frag16(kr + 32 + kb0, kr + 32 + kb0 + 16);
        v8f z = zero8();
        z = wmma_bf16(qa0, kf0, z);
        z = wmma_bf16(qa1, kf1, z);
        s[nt] = z;
      }
      // scale + causal mask
      const int rowb = i0 + wid * 16 + lhalf * 8;
      const int colb = jbase + l16;
      #pragma unroll
      for (int nt = 0; nt < 4; ++nt) {
        int col = colb + nt * 16;
        #pragma unroll
        for (int r = 0; r < 8; ++r) {
          float v = s[nt][r] * 0.125f;
          if (col > rowb + r) v = -3.0e38f;
          s[nt][r] = v;
        }
      }
      // online softmax: row reductions across the 16-lane half via shfl_xor
      float alpha[8];
      #pragma unroll
      for (int r = 0; r < 8; ++r) {
        float mx = fmaxf(fmaxf(s[0][r], s[1][r]), fmaxf(s[2][r], s[3][r]));
        #pragma unroll
        for (int off = 1; off < 16; off <<= 1)
          mx = fmaxf(mx, __shfl_xor(mx, off, 32));
        float mnew = fmaxf(mrow[r], mx);
        float a = __expf(mrow[r] - mnew);
        mrow[r] = mnew;
        float sum = 0.0f;
        #pragma unroll
        for (int nt = 0; nt < 4; ++nt) {
          float p = __expf(s[nt][r] - mnew);
          s[nt][r] = p;
          sum += p;
        }
        #pragma unroll
        for (int off = 1; off < 16; off <<= 1)
          sum += __shfl_xor(sum, off, 32);
        lrow[r] = lrow[r] * a + sum;
        alpha[r] = a;
      }
      #pragma unroll
      for (int dt = 0; dt < 4; ++dt)
        #pragma unroll
        for (int r = 0; r < 8; ++r)
          o[dt][r] *= alpha[r];
      // P: C-layout -> A-layout via per-wave LDS scratch (wave-private, in-order DS)
      bf16_t* Pw = Ps + wid * 16 * 72;
      #pragma unroll
      for (int nt = 0; nt < 4; ++nt)
        #pragma unroll
        for (int r = 0; r < 8; ++r)
          Pw[(lhalf * 8 + r) * 72 + nt * 16 + l16] = (bf16_t)s[nt][r];
      v16bf pa0 = frag16(&Pw[l16 * 72 + kb0],      &Pw[l16 * 72 + kb0 + 16]);
      v16bf pa1 = frag16(&Pw[l16 * 72 + 32 + kb0], &Pw[l16 * 72 + 32 + kb0 + 16]);
      // O += P * V
      #pragma unroll
      for (int dt = 0; dt < 4; ++dt) {
        const bf16_t* vr = &VtT[(dt * 16 + l16) * 72];
        v16bf vf0 = frag16(vr + kb0,      vr + kb0 + 16);
        v16bf vf1 = frag16(vr + 32 + kb0, vr + 32 + kb0 + 16);
        o[dt] = wmma_bf16(pa0, vf0, o[dt]);
        o[dt] = wmma_bf16(pa1, vf1, o[dt]);
      }
    }
    __syncthreads();
  }

  // normalize + write O in [B,T,C] bf16 (feeds final async-staged projection GEMM)
  #pragma unroll
  for (int r = 0; r < 8; ++r) {
    float inv = 1.0f / lrow[r];
    int t = i0 + wid * 16 + lhalf * 8 + r;
    #pragma unroll
    for (int dt = 0; dt < 4; ++dt) {
      int c = h * 64 + dt * 16 + l16;
      O[((size_t)b * TDIM + t) * CDIM + c] = (bf16_t)(o[dt][r] * inv);
    }
  }
}

// ---------------------------------------------------------------------------
extern "C" void kernel_launch(void* const* d_in, const int* in_sizes, int n_in,
                              void* d_out, int out_size, void* d_ws, size_t ws_size,
                              hipStream_t stream) {
  (void)in_sizes; (void)n_in; (void)out_size; (void)ws_size;
  const float* x  = (const float*)d_in[0];
  const float* wq = (const float*)d_in[1];
  const float* wk = (const float*)d_in[2];
  const float* wv = (const float*)d_in[3];
  const float* wp = (const float*)d_in[4];

  char* ws = (char*)d_ws;
  const size_t NX = (size_t)BDIM * TDIM * CDIM;              // 8.39M elems
  const size_t NW = (size_t)CDIM * CDIM;                     // 1.05M elems
  bf16_t* xb  = (bf16_t*)(ws);                               // 16 MB
  bf16_t* wqb = (bf16_t*)(ws + NX * 2);                      // 2 MB each
  bf16_t* wkb = wqb + NW;
  bf16_t* wvb = wkb + NW;
  bf16_t* wpb = wvb + NW;
  bf16_t* qb  = wpb + NW;                                    // 16 MB each (B,H,T,D)
  bf16_t* kb  = qb + NX;
  bf16_t* vb  = kb + NX;
  bf16_t* obb = vb + NX;                                     // 16 MB (B,T,C)
  float*  ctab = (float*)(obb + NX);
  float*  stab = ctab + TDIM * 32;

  cvt_kernel<<<(int)(NX / 1024), 256, 0, stream>>>(x,  xb,  (int)NX);
  cvt_kernel<<<(int)(NW / 1024), 256, 0, stream>>>(wq, wqb, (int)NW);
  cvt_kernel<<<(int)(NW / 1024), 256, 0, stream>>>(wk, wkb, (int)NW);
  cvt_kernel<<<(int)(NW / 1024), 256, 0, stream>>>(wv, wvb, (int)NW);
  cvt_kernel<<<(int)(NW / 1024), 256, 0, stream>>>(wp, wpb, (int)NW);
  rope_tab_kernel<<<256, 256, 0, stream>>>(ctab, stab);

  dim3 ggrid(CDIM / 128, (BDIM * TDIM) / 128);                // (8, 64)
  gemm_rope_kernel<<<ggrid, 256, 0, stream>>>(xb, wqb, nullptr, qb, ctab, stab, 1);
  gemm_rope_kernel<<<ggrid, 256, 0, stream>>>(xb, wkb, nullptr, kb, ctab, stab, 1);
  gemm_rope_kernel<<<ggrid, 256, 0, stream>>>(xb, wvb, nullptr, vb, ctab, stab, 2);

  flash_kernel<<<dim3(TDIM / 128, BDIM * NHEAD), 256, 0, stream>>>(qb, kb, vb, obb);

  gemm_rope_kernel<<<ggrid, 256, 0, stream>>>(obb, wpb, (float*)d_out, nullptr, ctab, stab, 0);
}
